// Discriminator_61332132987171
// MI455X (gfx1250) — matrix operand order repaired and checked by reading, not compiled, not run on hardware
//
#include <hip/hip_runtime.h>
#include <cstddef>
#include <cstdint>

// ---------------------------------------------------------------------------
// CDNA5 (gfx1250) wave32 WMMA implementation of the SN-GAN VQ discriminator.
// Heavy convs + VQ distances run on v_wmma_f32_16x16x32_f16 (f32 accum).
// Activations kept f16 NHWC; weight/codebook panels staged in LDS (padded,
// bank-conflict-free), shared by all 8 waves of a workgroup; 1 A-load feeds
// 4 WMMAs.  Panel staging uses GLOBAL_LOAD_ASYNC_TO_LDS (ASYNCcnt) when the
// toolchain exposes the builtin, else a sync VGPR round-trip.
// Hot kernels are templated on (H, W, KDIM, RELU) so all index math is
// shifts/masks instead of runtime division.
// ---------------------------------------------------------------------------

typedef _Float16 half8 __attribute__((ext_vector_type(8)));
typedef _Float16 v16h  __attribute__((ext_vector_type(16)));
typedef float    v8f   __attribute__((ext_vector_type(8)));
typedef int      i32x4 __attribute__((vector_size(16)));   // matches builtin param

#if defined(__has_builtin)
#  if __has_builtin(__builtin_amdgcn_global_load_async_to_lds_b128)
#    define USE_ASYNC_LDS 1
#  endif
#endif
#ifndef USE_ASYNC_LDS
#  define USE_ASYNC_LDS 0
#endif

__device__ __forceinline__ void stage16(const _Float16* g, _Float16* l) {
#if USE_ASYNC_LDS
  void* gp = (void*)const_cast<_Float16*>(g);
  __builtin_amdgcn_global_load_async_to_lds_b128(
      (__attribute__((address_space(1))) i32x4*)gp,
      (__attribute__((address_space(3))) i32x4*)(void*)l,
      0, 0);
#else
  *(half8*)l = *(const half8*)g;
#endif
}
__device__ __forceinline__ void stage_fence() {
#if USE_ASYNC_LDS
#  if __has_builtin(__builtin_amdgcn_s_wait_asynccnt)
  __builtin_amdgcn_s_wait_asynccnt(0);
#  else
  asm volatile("s_wait_asynccnt 0x0" ::: "memory");
#  endif
#endif
}

__device__ __forceinline__ v16h hzero16() {
  v16h z;
#pragma unroll
  for (int i = 0; i < 16; ++i) z[i] = (_Float16)0.0f;
  return z;
}
__device__ __forceinline__ v8f fzero8() {
  v8f z;
#pragma unroll
  for (int i = 0; i < 8; ++i) z[i] = 0.0f;
  return z;
}
__device__ __forceinline__ v16h cat8(half8 lo, half8 hi) {
  return __builtin_shufflevector(lo, hi, 0,1,2,3,4,5,6,7,8,9,10,11,12,13,14,15);
}
__device__ __forceinline__ v16h relu16(v16h a) {
#pragma unroll
  for (int i = 0; i < 16; ++i) a[i] = a[i] > (_Float16)0.0f ? a[i] : (_Float16)0.0f;
  return a;
}

// LDS panel: 64 rows (cout/code) x 256 halves, padded to 264 halves so the
// 16-lane fragment reads land on distinct bank groups (row stride = 132
// dwords -> lane i starts at bank (4*i + c) % 64, conflict-free).
#define PROW 264

// ---------------------------------------------------------------------------
// Spectral norm: sigma = ||W * normalize(W^T * ones)||  (1 power iteration)
// ---------------------------------------------------------------------------
__global__ void sn_sigma_kernel(const float* __restrict__ w, int O, int K,
                                float* __restrict__ sig) {
  __shared__ float shv[2304];   // max K = 256*9
  __shared__ float red[256];
  const int tid = threadIdx.x;
  for (int j = tid; j < K; j += 256) {
    float c = 0.0f;
    for (int i = 0; i < O; ++i) c += w[(size_t)i * K + j];
    shv[j] = c;
  }
  __syncthreads();
  float loc = 0.0f;
  for (int j = tid; j < K; j += 256) loc += shv[j] * shv[j];
  red[tid] = loc; __syncthreads();
  for (int s = 128; s > 0; s >>= 1) { if (tid < s) red[tid] += red[tid + s]; __syncthreads(); }
  const float nrm = sqrtf(red[0]) + 1e-8f;
  __syncthreads();
  for (int j = tid; j < K; j += 256) shv[j] /= nrm;
  __syncthreads();
  loc = 0.0f;
  for (int i = tid; i < O; i += 256) {
    const float* row = w + (size_t)i * K;
    float s = 0.0f;
    for (int j = 0; j < K; ++j) s += row[j] * shv[j];
    loc += s * s;
  }
  red[tid] = loc; __syncthreads();
  for (int s = 128; s > 0; s >>= 1) { if (tid < s) red[tid] += red[tid + s]; __syncthreads(); }
  if (tid == 0) sig[0] = sqrtf(red[0]);
}

__global__ void scale_f32_kernel(const float* __restrict__ w, const float* __restrict__ sig,
                                 float* __restrict__ out, int n) {
  int i = blockIdx.x * blockDim.x + threadIdx.x;
  if (i < n) out[i] = w[i] / sig[0];
}

// [cout][cin][ky][kx] f32 -> wt[(ky*3+kx)][cout][cin] f16 (cin contiguous)
__global__ void reorder_w3_kernel(const float* __restrict__ w, const float* __restrict__ sig,
                                  _Float16* __restrict__ wt) {
  int i = blockIdx.x * blockDim.x + threadIdx.x;
  if (i >= 256 * 256 * 9) return;
  int cout = i / (256 * 9);
  int rem  = i % (256 * 9);
  int cin  = rem / 9;
  int t    = rem % 9;
  wt[((size_t)(t * 256 + cout)) * 256 + cin] = (_Float16)(w[i] / sig[0]);
}

__global__ void reorder_w1_kernel(const float* __restrict__ w, const float* __restrict__ sig,
                                  _Float16* __restrict__ wt) {
  int i = blockIdx.x * blockDim.x + threadIdx.x;
  if (i < 256 * 256) wt[i] = (_Float16)(w[i] / sig[0]);
}

// codebook [256][1024] f32 -> cb16[1024][256] f16 + enorm[1024]
__global__ void prep_cb_kernel(const float* __restrict__ cb, _Float16* __restrict__ cb16,
                               float* __restrict__ enorm) {
  int k = blockIdx.x * blockDim.x + threadIdx.x;
  if (k >= 1024) return;
  float e = 0.0f;
  for (int c = 0; c < 256; ++c) {
    float v = cb[(size_t)c * 1024 + k];
    cb16[(size_t)k * 256 + c] = (_Float16)v;
    e += v * v;
  }
  enorm[k] = e;
}

__global__ void zero_kernel(float* __restrict__ p, int n) {
  int i = blockIdx.x * blockDim.x + threadIdx.x;
  if (i < n) p[i] = 0.0f;
}

// ---------------------------------------------------------------------------
// conv0: 3->256, 3x3, pad 1, 32x32; out = relu(conv+bias) as f16 NHWC.
// ---------------------------------------------------------------------------
__global__ void conv0_kernel(const float* __restrict__ x, const float* __restrict__ wn,
                             const float* __restrict__ bias, _Float16* __restrict__ out) {
  int idx = blockIdx.x * blockDim.x + threadIdx.x;
  if (idx >= 64 * 1024 * 256) return;
  int cout = idx & 255;
  int pix  = (idx >> 8) & 1023;
  int b    = idx >> 18;
  int py = pix >> 5, px = pix & 31;
  float acc = bias[cout];
  for (int cin = 0; cin < 3; ++cin)
    for (int ky = 0; ky < 3; ++ky) {
      int ys = py + ky - 1; if (ys < 0 || ys >= 32) continue;
      for (int kx = 0; kx < 3; ++kx) {
        int xs = px + kx - 1; if (xs < 0 || xs >= 32) continue;
        acc += x[((size_t)(b * 3 + cin) * 1024) + ys * 32 + xs] *
               wn[((size_t)(cout * 3 + cin) * 9) + ky * 3 + kx];
      }
    }
  out[idx] = (_Float16)(acc > 0.0f ? acc : 0.0f);
}

// ---------------------------------------------------------------------------
// Generic 256->256 conv (KDIM=3 pad 1, or KDIM=1) on WMMA, compile-time dims.
// Block = 8 waves; each wave owns one 16-pixel tile, all share one 64-cout
// group.  Per tap: cooperative (async) LDS stage of the 64x256 weight panel
// (32KB), then each k-chunk does 1 A-load -> 4 WMMAs.
// ---------------------------------------------------------------------------
template <int H, int W, int KDIM, int RELU>
__global__ __launch_bounds__(256) void conv_wmma_kernel(
    const _Float16* __restrict__ act, const _Float16* __restrict__ wt,
    const float* __restrict__ bias, _Float16* __restrict__ out) {
  __shared__ _Float16 bpanel[64][PROW];
  constexpr int HW = H * W;
  constexpr int pixTiles = HW >> 4;
  const int cg    = blockIdx.x & 3;          // cout group (64 couts)
  const int ftb   = (blockIdx.x >> 2) << 3;  // 8 flat pixel tiles per block
  const int wave  = threadIdx.x >> 5;
  const int lane  = threadIdx.x & 31;
  const int ft    = ftb + wave;              // flat (b, pixel-tile)
  const int b     = ft / pixTiles;           // pixTiles is a power of two
  const int pix_base = (ft % pixTiles) << 4;
  const int cout0 = cg << 6;

  const int l15   = lane & 15;
  const int koffA = (lane < 16) ? 0 : 8;     // A: 16-bit 16x32 layout
  const int kkB   = (lane < 16) ? 0 : 8;     // B: 16-bit 32x16 layout
  const int p  = pix_base + l15;
  const int py = p / W, px = p % W;          // W is a power of two

  v8f acc[4];
#pragma unroll
  for (int g = 0; g < 4; ++g) acc[g] = fzero8();

  constexpr int rad  = KDIM >> 1;
  constexpr int taps = KDIM * KDIM;
  for (int tap = 0; tap < taps; ++tap) {
    // --- stage wt[tap][cout0 .. cout0+63][0..255] into LDS (all 256 threads)
    __syncthreads();
#pragma unroll
    for (int it = 0; it < 8; ++it) {
      int idx  = threadIdx.x + it * 256;     // half8 index, 2048 total
      int row  = idx >> 5;                   // 32 half8 per 256-half row
      int col8 = idx & 31;
      stage16(wt + ((size_t)(tap * 256 + cout0 + row)) * 256 + col8 * 8,
              &bpanel[row][col8 * 8]);
    }
    stage_fence();
    __syncthreads();

    const int dy = tap / KDIM - rad, dx = tap % KDIM - rad;
    const int ys = py + dy, xs = px + dx;
    const bool valid = (ys >= 0) && (ys < H) && (xs >= 0) && (xs < W);
    const size_t abase = (size_t)(b * HW + ys * W + xs) * 256;
#pragma unroll
    for (int kb = 0; kb < 256; kb += 32) {
      v16h afrag;
      if (valid) {
        const half8* ap = (const half8*)(act + abase + kb + koffA);
        afrag = cat8(ap[0], ap[2]);
        if (RELU) afrag = relu16(afrag);
      } else {
        afrag = hzero16();
      }
#pragma unroll
      for (int g = 0; g < 4; ++g) {
        const half8* bp = (const half8*)&bpanel[g * 16 + l15][kb + kkB];
        v16h bfrag = cat8(bp[0], bp[2]);
        acc[g] = __builtin_amdgcn_wmma_f32_16x16x32_f16(false, afrag, false, bfrag,
                                                        (short)0, acc[g], false, false);
      }
    }
  }
  const int Moff = (lane < 16) ? 0 : 8;
#pragma unroll
  for (int g = 0; g < 4; ++g) {
    const int cn = cout0 + g * 16 + l15;
    const float bv = bias[cn];
#pragma unroll
    for (int r = 0; r < 8; ++r) {
      float v = acc[g][r] + bv;
      out[((size_t)(b * HW + pix_base + r + Moff)) * 256 + cn] = (_Float16)v;
    }
  }
}

// ---------------------------------------------------------------------------
// VQ quantize: block = 8 waves x 16 vectors; codebook staged 64 codes at a
// time in LDS (shared by all waves); WMMA x.e; argmin via 16-lane butterfly;
// SSE((q-x)^2) per batch via atomics; optional histogram for perplexity.
// ---------------------------------------------------------------------------
__global__ __launch_bounds__(256) void quantize_kernel(
    const _Float16* __restrict__ act, const _Float16* __restrict__ cb16,
    const float* __restrict__ enorm, float* __restrict__ sse,
    float* __restrict__ counts, int HW) {
  __shared__ _Float16 cpanel[64][PROW];
  __shared__ float enpanel[64];
  const int wave  = threadIdx.x >> 5;
  const int lane  = threadIdx.x & 31;
  const int vbase = (blockIdx.x * 8 + wave) << 4;
  const int l15   = lane & 15;
  const int koffA = (lane < 16) ? 0 : 8;
  const int kkB   = (lane < 16) ? 0 : 8;
  const int Moff  = (lane < 16) ? 0 : 8;
  const size_t abase = (size_t)(vbase + l15) * 256;

  float best[8]; int bidx[8];
#pragma unroll
  for (int r = 0; r < 8; ++r) { best[r] = 3.4e38f; bidx[r] = 0; }

  for (int cgrp = 0; cgrp < 16; ++cgrp) {    // 16 stages x 64 codes
    __syncthreads();
#pragma unroll
    for (int it = 0; it < 8; ++it) {
      int idx  = threadIdx.x + it * 256;
      int row  = idx >> 5;
      int col8 = idx & 31;
      stage16(cb16 + ((size_t)(cgrp * 64 + row)) * 256 + col8 * 8,
              &cpanel[row][col8 * 8]);
    }
    stage_fence();
    if (threadIdx.x < 64) enpanel[threadIdx.x] = enorm[cgrp * 64 + threadIdx.x];
    __syncthreads();

    for (int sub = 0; sub < 4; ++sub) {      // 4 code tiles of 16
      v8f acc = fzero8();
#pragma unroll
      for (int kb = 0; kb < 256; kb += 32) {
        const half8* ap = (const half8*)(act + abase + kb + koffA);
        v16h afrag = cat8(ap[0], ap[2]);
        const half8* bp = (const half8*)&cpanel[sub * 16 + l15][kb + kkB];
        v16h bfrag = cat8(bp[0], bp[2]);
        acc = __builtin_amdgcn_wmma_f32_16x16x32_f16(false, afrag, false, bfrag,
                                                     (short)0, acc, false, false);
      }
      const int   code = cgrp * 64 + sub * 16 + l15;
      const float en   = enpanel[sub * 16 + l15];
#pragma unroll
      for (int r = 0; r < 8; ++r) {
        float s = en - 2.0f * acc[r];        // ||e||^2 - 2 x.e  (||x||^2 const)
        int   c = code;
        for (int m = 1; m < 16; m <<= 1) {   // min over 16 codes (per half)
          float so = __shfl_xor(s, m, 32);
          int   co = __shfl_xor(c, m, 32);
          if (so < s || (so == s && co < c)) { s = so; c = co; }
        }
        if (s < best[r]) { best[r] = s; bidx[r] = c; }
      }
    }
  }
  // SSE(q - x): 16 lanes of each half split the 256 channels
  const int c0 = l15 * 16;
#pragma unroll
  for (int r = 0; r < 8; ++r) {
    const int v = vbase + r + Moff;
    const _Float16* xv = act  + (size_t)v * 256 + c0;
    const _Float16* qv = cb16 + (size_t)bidx[r] * 256 + c0;
    float s = 0.0f;
#pragma unroll
    for (int i = 0; i < 16; ++i) { float d = (float)qv[i] - (float)xv[i]; s += d * d; }
    for (int m = 1; m < 16; m <<= 1) s += __shfl_xor(s, m, 32);
    if (l15 == 0) {
      atomicAdd(&sse[v / HW], s);
      if (counts) atomicAdd(&counts[bidx[r]], 1.0f);
    }
  }
}

// ---------------------------------------------------------------------------
// Block-0 tail: h0 = avgpool2(y0b) + conv1x1(avgpool2(x), wsc) + bsc
// ---------------------------------------------------------------------------
__global__ void pool_add0_kernel(const _Float16* __restrict__ y0b, const float* __restrict__ x,
                                 const float* __restrict__ wscn, const float* __restrict__ bsc,
                                 _Float16* __restrict__ h0) {
  int idx = blockIdx.x * blockDim.x + threadIdx.x;
  if (idx >= 64 * 256 * 256) return;
  int c = idx & 255, opix = (idx >> 8) & 255, b = idx >> 16;
  int oy = opix >> 4, ox = opix & 15;
  float acc = 0.0f;
  for (int i = 0; i < 2; ++i)
    for (int j = 0; j < 2; ++j) {
      int p = (2 * oy + i) * 32 + (2 * ox + j);
      acc += (float)y0b[((size_t)(b * 1024 + p)) * 256 + c];
    }
  acc *= 0.25f;
  float sc = bsc[c];
  for (int cin = 0; cin < 3; ++cin) {
    float xp = 0.0f;
    for (int i = 0; i < 2; ++i)
      for (int j = 0; j < 2; ++j) {
        int p = (2 * oy + i) * 32 + (2 * ox + j);
        xp += x[((size_t)(b * 3 + cin) * 1024) + p];
      }
    sc += wscn[c * 3 + cin] * 0.25f * xp;
  }
  h0[idx] = (_Float16)(acc + sc);
}

// h1 = avgpool2(y1b) + avgpool2(sc1), NHWC 16x16 -> 8x8
__global__ void pool_add1_kernel(const _Float16* __restrict__ ym, const _Float16* __restrict__ ysc,
                                 _Float16* __restrict__ h1) {
  int idx = blockIdx.x * blockDim.x + threadIdx.x;
  if (idx >= 64 * 64 * 256) return;
  int c = idx & 255, opix = (idx >> 8) & 63, b = idx >> 14;
  int oy = opix >> 3, ox = opix & 7;
  float a = 0.0f, s = 0.0f;
  for (int i = 0; i < 2; ++i)
    for (int j = 0; j < 2; ++j) {
      int p = (2 * oy + i) * 16 + (2 * ox + j);
      size_t o = ((size_t)(b * 256 + p)) * 256 + c;
      a += (float)ym[o]; s += (float)ysc[o];
    }
  h1[idx] = (_Float16)(0.25f * (a + s));
}

__global__ void add_kernel(const _Float16* __restrict__ a, const _Float16* __restrict__ b,
                           _Float16* __restrict__ o, int n) {
  int i = blockIdx.x * blockDim.x + threadIdx.x;
  if (i < n) o[i] = (_Float16)((float)a[i] + (float)b[i]);
}

// hf[b][c] = sum over 64 pixels of relu(h3)
__global__ void hf_kernel(const _Float16* __restrict__ h3, float* __restrict__ hf) {
  int idx = blockIdx.x * blockDim.x + threadIdx.x;
  if (idx >= 64 * 256) return;
  int c = idx & 255, b = idx >> 8;
  float s = 0.0f;
  for (int p = 0; p < 64; ++p) {
    float v = (float)h3[((size_t)(b * 64 + p)) * 256 + c];
    s += v > 0.0f ? v : 0.0f;
  }
  hf[idx] = s;
}

__global__ void head_kernel(const float* __restrict__ hf, const float* __restrict__ linn,
                            const float* __restrict__ lin_b, const float* __restrict__ embn,
                            const int* __restrict__ y, float* __restrict__ dout) {
  int b = blockIdx.x * blockDim.x + threadIdx.x;
  if (b >= 64) return;
  float s = lin_b[0];
  int cls = y[b];
  for (int c = 0; c < 256; ++c) {
    float h = hf[b * 256 + c];
    s += h * linn[c] + h * embn[(size_t)cls * 256 + c];
  }
  dout[b] = s;
}

__global__ void quantfin_kernel(const float* __restrict__ sse, float* __restrict__ dout) {
  int b = blockIdx.x * blockDim.x + threadIdx.x;
  if (b >= 64) return;
  dout[64 + b] = 0.5f * (sse[b] / 65536.0f + sse[64 + b] / 16384.0f +
                         sse[128 + b] / 16384.0f + sse[192 + b] / 16384.0f);
}

__global__ void ppl_kernel(const float* __restrict__ counts, float* __restrict__ dout) {
  __shared__ float red[256];
  int tid = threadIdx.x;
  float s = 0.0f;
  for (int k = tid; k < 1024; k += 256) {
    float p = counts[k] / 4096.0f;
    s += p * logf(p + 1e-10f);
  }
  red[tid] = s; __syncthreads();
  for (int st = 128; st > 0; st >>= 1) { if (tid < st) red[tid] += red[tid + st]; __syncthreads(); }
  if (tid == 0) dout[128] = expf(-red[0]);
}

// ---------------------------------------------------------------------------
extern "C" void kernel_launch(void* const* d_in, const int* in_sizes, int n_in,
                              void* d_out, int out_size, void* d_ws, size_t ws_size,
                              hipStream_t stream) {
  (void)in_sizes; (void)n_in; (void)out_size; (void)ws_size;
  const float* x     = (const float*)d_in[0];
  const int*   y     = (const int*)d_in[1];
  const float* b0_w1 = (const float*)d_in[2];  const float* b0_b1 = (const float*)d_in[3];
  const float* b0_w2 = (const float*)d_in[4];  const float* b0_b2 = (const float*)d_in[5];
  const float* b0_wsc= (const float*)d_in[6];  const float* b0_bsc= (const float*)d_in[7];
  const float* b1_w1 = (const float*)d_in[8];  const float* b1_b1 = (const float*)d_in[9];
  const float* b1_w2 = (const float*)d_in[10]; const float* b1_b2 = (const float*)d_in[11];
  const float* b1_wsc= (const float*)d_in[12]; const float* b1_bsc= (const float*)d_in[13];
  const float* b2_w1 = (const float*)d_in[14]; const float* b2_b1 = (const float*)d_in[15];
  const float* b2_w2 = (const float*)d_in[16]; const float* b2_b2 = (const float*)d_in[17];
  const float* b3_w1 = (const float*)d_in[18]; const float* b3_b1 = (const float*)d_in[19];
  const float* b3_w2 = (const float*)d_in[20]; const float* b3_b2 = (const float*)d_in[21];
  const float* cbs[4] = {(const float*)d_in[22], (const float*)d_in[23],
                         (const float*)d_in[24], (const float*)d_in[25]};
  const float* lin_w = (const float*)d_in[26]; const float* lin_b = (const float*)d_in[27];
  const float* emb_w = (const float*)d_in[28];
  float* dout = (float*)d_out;

  char* ws = (char*)d_ws;
  size_t off = 0;
  auto alloc = [&](size_t bytes) -> void* {
    void* p = ws + off;
    off = (off + bytes + 255) & ~(size_t)255;
    return p;
  };
  float* sig   = (float*)alloc(12 * 4);
  float* b0w1n = (float*)alloc(6912 * 4);
  float* wscn  = (float*)alloc(768 * 4);
  float* linn  = (float*)alloc(256 * 4);
  float* embn  = (float*)alloc(25600 * 4);
  _Float16* wt_b0w2 = (_Float16*)alloc(589824 * 2);
  _Float16* wt_b1w1 = (_Float16*)alloc(589824 * 2);
  _Float16* wt_b1w2 = (_Float16*)alloc(589824 * 2);
  _Float16* wt_b2w1 = (_Float16*)alloc(589824 * 2);
  _Float16* wt_b2w2 = (_Float16*)alloc(589824 * 2);
  _Float16* wt_b3w1 = (_Float16*)alloc(589824 * 2);
  _Float16* wt_b3w2 = (_Float16*)alloc(589824 * 2);
  _Float16* wt_sc1  = (_Float16*)alloc(65536 * 2);
  _Float16* cb16s[4]; float* enorms[4];
  for (int i = 0; i < 4; ++i) {
    cb16s[i]  = (_Float16*)alloc(262144 * 2);
    enorms[i] = (float*)alloc(1024 * 4);
  }
  _Float16* y0a = (_Float16*)alloc((size_t)16777216 * 2);
  _Float16* y0b = (_Float16*)alloc((size_t)16777216 * 2);
  _Float16* h0  = (_Float16*)alloc((size_t)4194304 * 2);
  _Float16* y1a = (_Float16*)alloc((size_t)4194304 * 2);
  _Float16* y1b = (_Float16*)alloc((size_t)4194304 * 2);
  _Float16* sc1 = (_Float16*)alloc((size_t)4194304 * 2);
  _Float16* h1  = (_Float16*)alloc((size_t)1048576 * 2);
  _Float16* y2a = (_Float16*)alloc((size_t)1048576 * 2);
  _Float16* y2b = (_Float16*)alloc((size_t)1048576 * 2);
  _Float16* h2  = (_Float16*)alloc((size_t)1048576 * 2);
  _Float16* y3a = (_Float16*)alloc((size_t)1048576 * 2);
  _Float16* y3b = (_Float16*)alloc((size_t)1048576 * 2);
  _Float16* h3  = (_Float16*)alloc((size_t)1048576 * 2);
  float* hf     = (float*)alloc(16384 * 4);
  float* sse    = (float*)alloc(256 * 4);
  float* counts = (float*)alloc(1024 * 4);

  // --- spectral-norm sigmas (12 weights) ---
  sn_sigma_kernel<<<1, 256, 0, stream>>>(b0_w1, 256, 27,   sig + 0);
  sn_sigma_kernel<<<1, 256, 0, stream>>>(b0_w2, 256, 2304, sig + 1);
  sn_sigma_kernel<<<1, 256, 0, stream>>>(b0_wsc,256, 3,    sig + 2);
  sn_sigma_kernel<<<1, 256, 0, stream>>>(b1_w1, 256, 2304, sig + 3);
  sn_sigma_kernel<<<1, 256, 0, stream>>>(b1_w2, 256, 2304, sig + 4);
  sn_sigma_kernel<<<1, 256, 0, stream>>>(b1_wsc,256, 256,  sig + 5);
  sn_sigma_kernel<<<1, 256, 0, stream>>>(b2_w1, 256, 2304, sig + 6);
  sn_sigma_kernel<<<1, 256, 0, stream>>>(b2_w2, 256, 2304, sig + 7);
  sn_sigma_kernel<<<1, 256, 0, stream>>>(b3_w1, 256, 2304, sig + 8);
  sn_sigma_kernel<<<1, 256, 0, stream>>>(b3_w2, 256, 2304, sig + 9);
  sn_sigma_kernel<<<1, 256, 0, stream>>>(lin_w, 1,   256,  sig + 10);
  sn_sigma_kernel<<<1, 256, 0, stream>>>(emb_w, 100, 256,  sig + 11);

  // --- normalize / reorder / convert weights ---
  scale_f32_kernel<<<(6912 + 255) / 256, 256, 0, stream>>>(b0_w1, sig + 0, b0w1n, 6912);
  scale_f32_kernel<<<(768 + 255) / 256, 256, 0, stream>>>(b0_wsc, sig + 2, wscn, 768);
  scale_f32_kernel<<<1, 256, 0, stream>>>(lin_w, sig + 10, linn, 256);
  scale_f32_kernel<<<(25600 + 255) / 256, 256, 0, stream>>>(emb_w, sig + 11, embn, 25600);
  const int n3 = 256 * 256 * 9;
  reorder_w3_kernel<<<n3 / 256, 256, 0, stream>>>(b0_w2, sig + 1, wt_b0w2);
  reorder_w3_kernel<<<n3 / 256, 256, 0, stream>>>(b1_w1, sig + 3, wt_b1w1);
  reorder_w3_kernel<<<n3 / 256, 256, 0, stream>>>(b1_w2, sig + 4, wt_b1w2);
  reorder_w3_kernel<<<n3 / 256, 256, 0, stream>>>(b2_w1, sig + 6, wt_b2w1);
  reorder_w3_kernel<<<n3 / 256, 256, 0, stream>>>(b2_w2, sig + 7, wt_b2w2);
  reorder_w3_kernel<<<n3 / 256, 256, 0, stream>>>(b3_w1, sig + 8, wt_b3w1);
  reorder_w3_kernel<<<n3 / 256, 256, 0, stream>>>(b3_w2, sig + 9, wt_b3w2);
  reorder_w1_kernel<<<256, 256, 0, stream>>>(b1_wsc, sig + 5, wt_sc1);
  for (int i = 0; i < 4; ++i)
    prep_cb_kernel<<<4, 256, 0, stream>>>(cbs[i], cb16s[i], enorms[i]);
  zero_kernel<<<1, 256, 0, stream>>>(sse, 256);
  zero_kernel<<<4, 256, 0, stream>>>(counts, 1024);

  auto quant = [&](const _Float16* act, int stage, int HW, float* cnts) {
    int blocks = 64 * HW / 128;               // 128 vectors per block
    quantize_kernel<<<blocks, 256, 0, stream>>>(act, cb16s[stage], enorms[stage],
                                                sse + stage * 64, cnts, HW);
  };
  // conv blocks = (B*HW/16 pixel-tiles / 8 per block) * 4 cout groups
  constexpr int CB32 = (64 * 1024 / 16 / 8) * 4;  // 2048
  constexpr int CB16 = (64 * 256  / 16 / 8) * 4;  // 512
  constexpr int CB8  = (64 * 64   / 16 / 8) * 4;  // 128

  // --- block 0 (preact=False, down) ---
  conv0_kernel<<<(64 * 1024 * 256) / 256, 256, 0, stream>>>(x, b0w1n, b0_b1, y0a); // relu fused
  conv_wmma_kernel<32,32,3,0><<<CB32, 256, 0, stream>>>(y0a, wt_b0w2, b0_b2, y0b);
  pool_add0_kernel<<<(64 * 256 * 256) / 256, 256, 0, stream>>>(y0b, x, wscn, b0_bsc, h0);
  quant(h0, 0, 256, nullptr);

  // --- block 1 (preact=True, down) ---
  conv_wmma_kernel<16,16,3,1><<<CB16, 256, 0, stream>>>(h0, wt_b1w1, b1_b1, y1a);
  conv_wmma_kernel<16,16,3,1><<<CB16, 256, 0, stream>>>(y1a, wt_b1w2, b1_b2, y1b);
  conv_wmma_kernel<16,16,1,0><<<CB16, 256, 0, stream>>>(h0, wt_sc1, b1_bsc, sc1);
  pool_add1_kernel<<<(64 * 64 * 256) / 256, 256, 0, stream>>>(y1b, sc1, h1);
  quant(h1, 1, 64, nullptr);

  // --- block 2 (preact=True, identity sc) ---
  conv_wmma_kernel<8,8,3,1><<<CB8, 256, 0, stream>>>(h1, wt_b2w1, b2_b1, y2a);
  conv_wmma_kernel<8,8,3,1><<<CB8, 256, 0, stream>>>(y2a, wt_b2w2, b2_b2, y2b);
  add_kernel<<<(64 * 64 * 256) / 256, 256, 0, stream>>>(y2b, h1, h2, 64 * 64 * 256);
  quant(h2, 2, 64, nullptr);

  // --- block 3 ---
  conv_wmma_kernel<8,8,3,1><<<CB8, 256, 0, stream>>>(h2, wt_b3w1, b3_b1, y3a);
  conv_wmma_kernel<8,8,3,1><<<CB8, 256, 0, stream>>>(y3a, wt_b3w2, b3_b2, y3b);
  add_kernel<<<(64 * 64 * 256) / 256, 256, 0, stream>>>(y3b, h2, h3, 64 * 64 * 256);
  quant(h3, 3, 64, counts);

  // --- head + losses ---
  hf_kernel<<<64, 256, 0, stream>>>(h3, hf);
  head_kernel<<<1, 64, 0, stream>>>(hf, linn, lin_b, embn, y, dout);
  quantfin_kernel<<<1, 64, 0, stream>>>(sse, dout);
  ppl_kernel<<<1, 256, 0, stream>>>(counts, dout);
}
